// AFTLocalAutoregressive_74629351735849
// MI455X (gfx1250) — compile-verified, matrix operand order checked by reading, not run on hardware
//
#include <hip/hip_runtime.h>
#include <hip/hip_bf16.h>

// Problem constants (fixed by the reference).
#define S   4096
#define Bq  2
#define D   512
#define WIN 128
#define MROWS (S * Bq)   // 8192
#define BD    (Bq * D)   // 1024

// ---------------- WMMA types ----------------
typedef __attribute__((ext_vector_type(16))) __bf16 v16bf;
typedef __attribute__((ext_vector_type(8)))  float  v8f;

__device__ __forceinline__ unsigned short f2bf(float f) {
    unsigned int u = __float_as_uint(f);
    u += 0x7FFFu + ((u >> 16) & 1u);   // round-to-nearest-even
    return (unsigned short)(u >> 16);
}

// ---------------------------------------------------------------------------
// fp32 -> bf16 pack (activations). Reads float4, writes ushort4 (8B).
// ---------------------------------------------------------------------------
__global__ __launch_bounds__(256) void cvt_bf16_kernel(
    const float* __restrict__ src, unsigned short* __restrict__ dst, int n4)
{
    int i = blockIdx.x * 256 + threadIdx.x;
    if (i < n4) {
        float4 f = ((const float4*)src)[i];
        ushort4 o;
        o.x = f2bf(f.x); o.y = f2bf(f.y); o.z = f2bf(f.z); o.w = f2bf(f.w);
        ((ushort4*)dst)[i] = o;
    }
}

// ---------------------------------------------------------------------------
// W[k][n] fp32 -> Wt[n][k] bf16  (32x32 LDS tile transpose; D x D matrices).
// One-time cost so the GEMM's B operand is row-contiguous -> async-copyable.
// ---------------------------------------------------------------------------
__global__ __launch_bounds__(256) void transpose_cvt_kernel(
    const float* __restrict__ W, unsigned short* __restrict__ Wt)
{
    __shared__ float tile[32][33];
    int c0 = blockIdx.x * 32, r0 = blockIdx.y * 32;
    int tx = threadIdx.x & 31, ty = threadIdx.x >> 5;  // 32 x 8
    #pragma unroll
    for (int i = 0; i < 32; i += 8)
        tile[ty + i][tx] = W[(size_t)(r0 + ty + i) * D + c0 + tx];
    __syncthreads();
    #pragma unroll
    for (int i = 0; i < 32; i += 8)
        Wt[(size_t)(c0 + ty + i) * D + r0 + tx] = f2bf(tile[tx][ty + i]);
}

// ---------------------------------------------------------------------------
// GEMM: C[M,N] = Abf[M,K] @ Wt[N,K]^T + bias[N]   (bf16 WMMA, fp32 accum).
// Block tile 128x128, K-step 32, double-buffered LDS filled with
// global_load_async_to_lds_b128 (ASYNCcnt). 8 wave32 waves as 4(M) x 2(N);
// each wave: 32x64 accumulator = 2x4 v_wmma_f32_16x16x32_bf16 per K-step.
// Both operands are stored [row][k] (k contiguous) so every tile row is a
// straight 64B copy and every fragment is one contiguous 32B LDS read.
// ---------------------------------------------------------------------------
#define BM 128
#define BN 128
#define BK 32
// per-buffer: A 128*32 bf16 = 8KB, B 128*32 bf16 = 8KB; x2 buffers = 32KB LDS

__device__ __forceinline__ void stage_tiles(
    const unsigned short* Ab, const unsigned short* Wt,
    unsigned sbase, int buf, int row0, int col0, int kk, int K, int tid)
{
    unsigned abase = sbase + (unsigned)buf * 16384u;
    unsigned bbase = abase + 8192u;
    #pragma unroll
    for (int it = 0; it < 2; ++it) {
        int idx = tid + it * 256;          // 0..511 -> 128 rows x 4 chunks of 8 bf16
        int r   = idx >> 2;
        int c8  = (idx & 3) << 3;
        unsigned ldsa  = abase + (unsigned)((r * BK + c8) * 2);
        unsigned voffa = (unsigned)(((row0 + r) * K + kk + c8) * 2);
        asm volatile("global_load_async_to_lds_b128 %0, %1, %2"
                     :: "v"(ldsa), "v"(voffa), "s"(Ab) : "memory");
        unsigned ldsb  = bbase + (unsigned)((r * BK + c8) * 2);
        unsigned voffb = (unsigned)(((col0 + r) * K + kk + c8) * 2);
        asm volatile("global_load_async_to_lds_b128 %0, %1, %2"
                     :: "v"(ldsb), "v"(voffb), "s"(Wt) : "memory");
    }
}

__global__ __launch_bounds__(256) void gemm_bias_bf16wmma(
    const unsigned short* __restrict__ Ab, const unsigned short* __restrict__ Wt,
    const float* __restrict__ bias, float* __restrict__ C,
    int M, int N, int K)
{
    __shared__ alignas(16) unsigned short smem[2 * 8192];

    const int tid   = threadIdx.x;
    const int lane  = tid & 31;
    const int wave  = tid >> 5;
    const int waveM = wave >> 1;       // 0..3
    const int waveN = wave & 1;        // 0..1
    const int row0  = blockIdx.y * BM;
    const int col0  = blockIdx.x * BN;
    const int l15   = lane & 15;
    const int kb    = (lane >> 4) * 16;  // contiguous k order used for A and B: lanes 16-31 hold K=16..31

    const unsigned sbase = (unsigned)(size_t)(void*)smem;

    v8f acc[2][4] = {};

    stage_tiles(Ab, Wt, sbase, 0, row0, col0, 0, K, tid);

    const int steps = K / BK;
    for (int s = 0; s < steps; ++s) {
        int buf = s & 1;
        asm volatile("s_wait_asynccnt 0x0" ::: "memory");
        __syncthreads();                                   // tile[buf] ready everywhere
        if (s + 1 < steps)
            stage_tiles(Ab, Wt, sbase, buf ^ 1, row0, col0, (s + 1) * BK, K, tid);

        const unsigned short* sA = smem + buf * 8192;
        const unsigned short* sB = sA + 4096;

        v16bf a[2], b[4];
        #pragma unroll
        for (int i = 0; i < 2; ++i)
            a[i] = *(const v16bf*)&sA[(waveM * 32 + i * 16 + l15) * BK + kb];
        #pragma unroll
        for (int j = 0; j < 4; ++j)
            b[j] = *(const v16bf*)&sB[(waveN * 64 + j * 16 + l15) * BK + kb];

        #pragma unroll
        for (int i = 0; i < 2; ++i)
            #pragma unroll
            for (int j = 0; j < 4; ++j)
                acc[i][j] = __builtin_amdgcn_wmma_f32_16x16x32_bf16(
                    false, a[i], false, b[j], (short)0, acc[i][j], false, false);
        __syncthreads();                                   // all reads of tile[buf] done
    }

    // Epilogue: VGPR e holds M=e (lanes 0-15) / M=e+8 (lanes 16-31), N=lane&15.
    const int rbase = (lane >> 4) * 8;
    #pragma unroll
    for (int i = 0; i < 2; ++i) {
        int rg0 = row0 + waveM * 32 + i * 16 + rbase;
        #pragma unroll
        for (int j = 0; j < 4; ++j) {
            int cg = col0 + waveN * 64 + j * 16 + l15;
            float bv = bias[cg];
            float* cp = C + (size_t)rg0 * N + cg;
            #pragma unroll
            for (int e = 0; e < 8; ++e)
                cp[(size_t)e * N] = acc[i][j][e] + bv;     // const offsets -> imm24
        }
    }
}

// ---------------------------------------------------------------------------
// colmax[j] = max_i weights[i][j]        (coalesced over j)
// ---------------------------------------------------------------------------
__global__ __launch_bounds__(256) void colmax_kernel(
    const float* __restrict__ weights, float* __restrict__ colmax)
{
    int j = blockIdx.x * blockDim.x + threadIdx.x;
    float cm = -3.402823466e38f;
    for (int i = 0; i < S; ++i) cm = fmaxf(cm, weights[(size_t)i * S + j]);
    colmax[j] = cm;
}

// ---------------------------------------------------------------------------
// m[e] = max( max_s k[s,e], max_s (k[s,e] + colmax[s]) ),  e = b*D+d
// ---------------------------------------------------------------------------
__global__ __launch_bounds__(256) void mmax_kernel(
    const float* __restrict__ kbuf, const float* __restrict__ colmax,
    float* __restrict__ mbuf)
{
    int e = blockIdx.x * blockDim.x + threadIdx.x;
    float m1 = -3.402823466e38f, m2 = -3.402823466e38f;
    for (int s = 0; s < S; ++s) {
        float kv = kbuf[s * BD + e];
        m1 = fmaxf(m1, kv);
        m2 = fmaxf(m2, kv + colmax[s]);
    }
    mbuf[e] = fmaxf(m1, m2);
}

// ---------------------------------------------------------------------------
// Shifted prefix sums: num_pre[t] = cumsum_s(exp(k-m)*v)[t-WIN] (0 for t<WIN).
// ---------------------------------------------------------------------------
__global__ __launch_bounds__(256) void prefix_kernel(
    const float* __restrict__ kbuf, const float* __restrict__ vbuf,
    const float* __restrict__ mbuf,
    float* __restrict__ numpre, float* __restrict__ denpre)
{
    int e = blockIdx.x * blockDim.x + threadIdx.x;
    float mv = mbuf[e];
    for (int s = 0; s < WIN; ++s) {
        numpre[s * BD + e] = 0.0f;
        denpre[s * BD + e] = 0.0f;
    }
    float cn = 0.0f, cd = 0.0f;
    for (int s = 0; s < S - WIN; ++s) {
        int idx = s * BD + e;
        float ex = __expf(kbuf[idx] - mv);
        cn += ex * vbuf[idx];
        cd += ex;
        int ti = (s + WIN) * BD + e;
        numpre[ti] = cn;
        denpre[ti] = cd;
    }
}

// ---------------------------------------------------------------------------
// 128-tap window + sigmoid gate; writes x directly as bf16 for the out-GEMM.
// One block per t; the 128 weight diagonals for this t staged in LDS.
// ---------------------------------------------------------------------------
__global__ __launch_bounds__(256) void window_kernel(
    const float* __restrict__ qb, const float* __restrict__ kbuf,
    const float* __restrict__ vbuf, const float* __restrict__ mbuf,
    const float* __restrict__ numpre, const float* __restrict__ denpre,
    const float* __restrict__ weights, unsigned short* __restrict__ xbf)
{
    __shared__ float wd[WIN];
    const int t   = blockIdx.x;
    const int tid = threadIdx.x;
    if (tid < WIN) {
        int c = t - tid;
        if (c < 0) c = 0;                       // clip(t-j, 0)
        wd[tid] = weights[(size_t)t * S + c];
    }
    __syncthreads();

    const int jmax = (t < WIN - 1) ? t : (WIN - 1);
    for (int e = tid; e < BD; e += 256) {
        float mv = mbuf[e];
        float num = 0.0f, den = 0.0f;
        for (int jv = 0; jv <= jmax; ++jv) {
            int idx = (t - jv) * BD + e;
            float ex = __expf(kbuf[idx] + wd[jv] - mv);
            num += ex * vbuf[idx];
            den += ex;
        }
        int ti = t * BD + e;
        float qv  = qb[ti];
        float sig = 1.0f / (1.0f + __expf(-qv));
        float x   = sig * (num + numpre[ti]) / (den + denpre[ti]);
        xbf[ti]   = f2bf(x);
    }
}

// ---------------------------------------------------------------------------
extern "C" void kernel_launch(void* const* d_in, const int* in_sizes, int n_in,
                              void* d_out, int out_size, void* d_ws, size_t ws_size,
                              hipStream_t stream)
{
    const float* query   = (const float*)d_in[0];
    const float* key     = (const float*)d_in[1];
    const float* value   = (const float*)d_in[2];
    const float* Wq      = (const float*)d_in[3];
    const float* bqp     = (const float*)d_in[4];
    const float* Wk      = (const float*)d_in[5];
    const float* bkp     = (const float*)d_in[6];
    const float* Wv      = (const float*)d_in[7];
    const float* bvp     = (const float*)d_in[8];
    const float* Wo      = (const float*)d_in[9];
    const float* bop     = (const float*)d_in[10];
    const float* weights = (const float*)d_in[11];
    // d_in[12] = s (window) — fixed at 128 by the reference.

    const size_t MBD = (size_t)MROWS * D;     // 4,194,304
    const size_t DD  = (size_t)D * D;         // 262,144

    // fp32 region
    float* ws     = (float*)d_ws;
    float* qf     = ws;                  // q projection (fp32)
    float* kf     = qf + MBD;
    float* vf     = kf + MBD;
    float* numpre = vf + MBD;
    float* denpre = numpre + MBD;
    float* cmax   = denpre + MBD;        // S floats
    float* mbuf   = cmax + S;            // BD floats
    // bf16 region (ushort), starts 16B-aligned (all prior sizes are multiples of 4 floats)
    unsigned short* bfr = (unsigned short*)(mbuf + BD);
    unsigned short* qib = bfr;           // bf16 copies of the inputs
    unsigned short* kib = qib + MBD;
    unsigned short* vib = kib + MBD;
    unsigned short* Wqt = vib + MBD;     // transposed bf16 weights [N][K]
    unsigned short* Wkt = Wqt + DD;
    unsigned short* Wvt = Wkt + DD;
    unsigned short* Wot = Wvt + DD;
    unsigned short* xbf = Wot + DD;      // gated output, bf16

    const int n4 = (int)(MBD / 4);
    dim3 tgrid(D / 32, D / 32);                 // (16,16)
    dim3 ggrid(D / BN, MROWS / BM);             // (4, 64)

    // 1) precision staging: activations -> bf16, weights -> transposed bf16
    cvt_bf16_kernel<<<n4 / 256, 256, 0, stream>>>(query, qib, n4);
    cvt_bf16_kernel<<<n4 / 256, 256, 0, stream>>>(key,   kib, n4);
    cvt_bf16_kernel<<<n4 / 256, 256, 0, stream>>>(value, vib, n4);
    transpose_cvt_kernel<<<tgrid, 256, 0, stream>>>(Wq, Wqt);
    transpose_cvt_kernel<<<tgrid, 256, 0, stream>>>(Wk, Wkt);
    transpose_cvt_kernel<<<tgrid, 256, 0, stream>>>(Wv, Wvt);
    transpose_cvt_kernel<<<tgrid, 256, 0, stream>>>(Wo, Wot);

    // 2) projections (bf16 WMMA, async-LDS staged)
    gemm_bias_bf16wmma<<<ggrid, 256, 0, stream>>>(qib, Wqt, bqp, qf, MROWS, D, D);
    gemm_bias_bf16wmma<<<ggrid, 256, 0, stream>>>(kib, Wkt, bkp, kf, MROWS, D, D);
    gemm_bias_bf16wmma<<<ggrid, 256, 0, stream>>>(vib, Wvt, bvp, vf, MROWS, D, D);

    // 3) AFT-local stabilized accumulation
    colmax_kernel<<<S / 256, 256, 0, stream>>>(weights, cmax);
    mmax_kernel<<<BD / 256, 256, 0, stream>>>(kf, cmax, mbuf);
    prefix_kernel<<<BD / 256, 256, 0, stream>>>(kf, vf, mbuf, numpre, denpre);
    window_kernel<<<S, 256, 0, stream>>>(qf, kf, vf, mbuf, numpre, denpre, weights, xbf);

    // 4) output projection
    gemm_bias_bf16wmma<<<ggrid, 256, 0, stream>>>(xbf, Wot, bop, (float*)d_out, MROWS, D, D);
}
